// SeparationHead_43971875176949
// MI455X (gfx1250) — matrix operand order repaired and checked by reading, not compiled, not run on hardware
//
#include <hip/hip_runtime.h>
#include <hip/hip_bf16.h>
#include <stdint.h>

// Problem constants (from reference): B,S,C,D,G
constexpr int NB = 64;    // batch
constexpr int NS = 512;   // separations
constexpr int NC = 4096;  // cubes per graph
constexpr int ND = 128;   // node dim
constexpr int NG = 128;   // global dim

typedef __attribute__((ext_vector_type(16))) _Float16 v16h;
typedef __attribute__((ext_vector_type(8)))  float    v8f;

union Frag16 {
  v16h v;
  uint4 q[2];
};

// Expand 4 mask bytes (each 0/1) in word w into two packed-f16 words using
// v_perm_b32: place byte pair at halfword positions, then *0x3C00 (f16 1.0).
__device__ __forceinline__ unsigned packPair0(unsigned w) {
  return __builtin_amdgcn_perm(0u, w, 0x0C010C00u) * 0x3C00u;
}
__device__ __forceinline__ unsigned packPair1(unsigned w) {
  return __builtin_amdgcn_perm(0u, w, 0x0C030C02u) * 0x3C00u;
}

// ---------------------------------------------------------------------------
// Prep 1: node_features f32 [B*C, D] -> f16 transposed [B][D][C] in workspace
// so WMMA B-fragments are two contiguous 16B loads per lane.
// ---------------------------------------------------------------------------
__global__ __launch_bounds__(256) void prep_transpose(const float* __restrict__ in,
                                                      _Float16* __restrict__ out) {
  __shared__ _Float16 tile[128 * 64];   // [n][k] for a 64-k slab
  const int b  = blockIdx.x >> 6;
  const int k0 = (blockIdx.x & 63) << 6;
  const int t  = threadIdx.x;
  const float* src = in + ((size_t)b * NC + k0) * ND;
#pragma unroll
  for (int i = 0; i < 32; ++i) {
    int idx = i * 256 + t;            // 0..8191, coalesced over rows
    int k = idx >> 7, n = idx & 127;
    tile[n * 64 + k] = (_Float16)src[idx];
  }
  __syncthreads();
  _Float16* dst = out + (size_t)b * ND * NC + k0;
#pragma unroll
  for (int i = 0; i < 32; ++i) {
    int idx = i * 256 + t;
    int n = idx >> 6, k = idx & 63;
    dst[(size_t)n * NC + k] = tile[idx];
  }
}

// ---------------------------------------------------------------------------
// Prep 2: W1 [128][256], W2 [64][128] f32 -> f16 (row-major, row = output n)
// ---------------------------------------------------------------------------
__global__ __launch_bounds__(256) void prep_weights(const float* __restrict__ W1,
                                                    const float* __restrict__ W2,
                                                    _Float16* __restrict__ W1h,
                                                    _Float16* __restrict__ W2h) {
  int idx = blockIdx.x * 256 + threadIdx.x;
  if (idx < 128 * 256) {
    W1h[idx] = (_Float16)W1[idx];
  } else {
    int j = idx - 128 * 256;
    if (j < 64 * 128) W2h[j] = (_Float16)W2[j];
  }
}

// ---------------------------------------------------------------------------
// Prep 3: gW1[b][n] = sum_k g[b][k] * W1[n][k] + b1[n]   (g-half of the MLP
// input is row-invariant, so it folds into a per-batch bias for h1)
// ---------------------------------------------------------------------------
__global__ __launch_bounds__(128) void prep_gw1(const float* __restrict__ gf,
                                                const float* __restrict__ W1,
                                                const float* __restrict__ b1,
                                                float* __restrict__ gW1b) {
  const int b = blockIdx.x, n = threadIdx.x;
  const float* g = gf + (size_t)b * NG;
  const float* w = W1 + (size_t)n * (NG + ND);   // first NG columns = g part
  float s = b1[n];
#pragma unroll 8
  for (int k = 0; k < NG; ++k) s += g[k] * w[k];
  gW1b[b * 128 + n] = s;
}

// ---------------------------------------------------------------------------
// Main fused kernel: masked-mean GEMM (WMMA f16) + 3-layer MLP (WMMA) + mask
// One workgroup = one (batch, 64-row s-tile). 8 waves, wave w owns N-tile w.
// K is staged 64 wide per barrier, double-buffered in LDS (1 barrier/step).
// ---------------------------------------------------------------------------
__global__ __launch_bounds__(256) void sep_head_main(
    const unsigned char* __restrict__ scm,   // [B,S,C] bool
    const unsigned char* __restrict__ svm,   // [B,S]   bool
    const _Float16* __restrict__ featsT,     // [B][D][C] f16
    const _Float16* __restrict__ W1h,        // [128][256] f16
    const _Float16* __restrict__ W2h,        // [64][128] f16
    const float* __restrict__ gW1b,          // [B][128] f32
    const float* __restrict__ b2,
    const float* __restrict__ W3,            // [64] f32
    const float* __restrict__ b3,
    float* __restrict__ out)                 // [B,S] f32
{
  __shared__ _Float16 aTile[2][64 * 64];  // double-buffered mask tile (f16)
  __shared__ _Float16 aggLds[64 * 128];   // agg (masked mean) f16
  __shared__ _Float16 h1Lds[64 * 128];
  __shared__ _Float16 h2Lds[64 * 64];
  __shared__ int cntLds[64];

  const int t    = threadIdx.x;
  const int lane = t & 31;
  const int w    = t >> 5;         // wave id 0..7 = N-tile
  const int ml   = lane & 15;
  const int hi   = lane >> 4;      // 0/1 (lane half)
  const int b    = blockIdx.x >> 3;
  const int s0   = (blockIdx.x & 7) << 6;

  if (t < 64) cntLds[t] = 0;

  // --- staging role: thread t feeds (row = t/4, 16 bytes at (t%4)*16) ---
  const int srow = t >> 2;
  const int skc  = (t & 3) << 4;
  const unsigned char* mrow = scm + ((size_t)(b * NS + s0 + srow)) * NC + skc;
  const int sbase = srow * 32 + (t & 3) * 8;   // u32 index into a 64x64 tile
  int cntPart = 0;

  // --- compute role (per documented 16-bit WMMA fragment layouts) ---
  const int nCol = (w << 4) + ml;        // output column of this lane
  const int kcA  = hi << 3;              // A frag: K chunks {kcA..+7, kcA+16..+23}
  const int kbB  = hi << 3;              // B frag: K chunks {kbB..+7, kbB+16..+23}
  const _Float16* bBase = featsT + (size_t)b * ND * NC + (size_t)nCol * NC + kbB;

  v8f acc[4] = {};                       // 4 M-subtiles x (16x16 f32)

  // prefetch k-step 0 (16 mask bytes per thread)
  unsigned long long qa = *(const unsigned long long*)mrow;
  unsigned long long qb = *(const unsigned long long*)(mrow + 8);
  __syncthreads();

  for (int kk = 0; kk < NC / 64; ++kk) {
    // ---- stage current 64-wide mask slab into LDS buffer kk&1 ----
    cntPart += __popcll(qa) + __popcll(qb);
    unsigned w0 = (unsigned)qa, w1 = (unsigned)(qa >> 32);
    unsigned w2 = (unsigned)qb, w3_ = (unsigned)(qb >> 32);
    uint4 st0, st1;
    st0.x = packPair0(w0); st0.y = packPair1(w0);
    st0.z = packPair0(w1); st0.w = packPair1(w1);
    st1.x = packPair0(w2); st1.y = packPair1(w2);
    st1.z = packPair0(w3_); st1.w = packPair1(w3_);
    unsigned* aU = (unsigned*)aTile[kk & 1];
    *(uint4*)&aU[sbase]     = st0;
    *(uint4*)&aU[sbase + 4] = st1;
    __syncthreads();                       // single barrier per K-step
    if (kk + 1 < NC / 64) {
      qa = *(const unsigned long long*)(mrow + (size_t)(kk + 1) * 64);
      qb = *(const unsigned long long*)(mrow + (size_t)(kk + 1) * 64 + 8);
    }

    // ---- compute: two 32-K substeps, 8 WMMAs per wave ----
    const _Float16* atb = aTile[kk & 1];
#pragma unroll
    for (int s = 0; s < 2; ++s) {
      Frag16 bf;
      const _Float16* bp = bBase + kk * 64 + s * 32;
      bf.q[0] = *(const uint4*)bp;
      bf.q[1] = *(const uint4*)(bp + 16);
#pragma unroll
      for (int m = 0; m < 4; ++m) {
        Frag16 af;
        const _Float16* ap = &atb[(m * 16 + ml) * 64 + s * 32 + kcA];
        af.q[0] = *(const uint4*)ap;
        af.q[1] = *(const uint4*)(ap + 16);
        acc[m] = __builtin_amdgcn_wmma_f32_16x16x32_f16(
            false, af.v, false, bf.v, (short)0, acc[m], false, false);
      }
    }
  }

  // finalize counts
  atomicAdd(&cntLds[srow], cntPart);
  __syncthreads();

  // agg = acc / max(cnt,1)  -> LDS f16  (C layout: VGPR r -> M=r+8*hi, N=ml)
#pragma unroll
  for (int m = 0; m < 4; ++m) {
#pragma unroll
    for (int r = 0; r < 8; ++r) {
      int row = m * 16 + r + (hi << 3);
      float v = acc[m][r] / fmaxf((float)cntLds[row], 1.0f);
      aggLds[row * 128 + nCol] = (_Float16)v;
    }
  }
  __syncthreads();

  // h1 = relu(agg @ W1_agg^T + gW1b)  : M=64, K=128, N=128
  v8f h1a[4] = {};
#pragma unroll
  for (int kk = 0; kk < 4; ++kk) {
    Frag16 bf;
    const _Float16* bp = W1h + (size_t)nCol * 256 + 128 + kk * 32 + kbB;
    bf.q[0] = *(const uint4*)bp;
    bf.q[1] = *(const uint4*)(bp + 16);
#pragma unroll
    for (int m = 0; m < 4; ++m) {
      Frag16 af;
      const _Float16* ap = &aggLds[(m * 16 + ml) * 128 + kk * 32 + kcA];
      af.q[0] = *(const uint4*)ap;
      af.q[1] = *(const uint4*)(ap + 16);
      h1a[m] = __builtin_amdgcn_wmma_f32_16x16x32_f16(
          false, af.v, false, bf.v, (short)0, h1a[m], false, false);
    }
  }
  {
    float bias = gW1b[b * 128 + nCol];
#pragma unroll
    for (int m = 0; m < 4; ++m)
#pragma unroll
      for (int r = 0; r < 8; ++r) {
        int row = m * 16 + r + (hi << 3);
        h1Lds[row * 128 + nCol] = (_Float16)fmaxf(h1a[m][r] + bias, 0.0f);
      }
  }
  __syncthreads();

  // h2 = relu(h1 @ W2^T + b2) : M=64, K=128, N=64  (waves 0..3)
  if (w < 4) {
    v8f h2a[4] = {};
#pragma unroll
    for (int kk = 0; kk < 4; ++kk) {
      Frag16 bf;
      const _Float16* bp = W2h + (size_t)nCol * 128 + kk * 32 + kbB;
      bf.q[0] = *(const uint4*)bp;
      bf.q[1] = *(const uint4*)(bp + 16);
#pragma unroll
      for (int m = 0; m < 4; ++m) {
        Frag16 af;
        const _Float16* ap = &h1Lds[(m * 16 + ml) * 128 + kk * 32 + kcA];
        af.q[0] = *(const uint4*)ap;
        af.q[1] = *(const uint4*)(ap + 16);
        h2a[m] = __builtin_amdgcn_wmma_f32_16x16x32_f16(
            false, af.v, false, bf.v, (short)0, h2a[m], false, false);
      }
    }
    float bias2 = b2[nCol];
#pragma unroll
    for (int m = 0; m < 4; ++m)
#pragma unroll
      for (int r = 0; r < 8; ++r) {
        int row = m * 16 + r + (hi << 3);
        h2Lds[row * 64 + nCol] = (_Float16)fmaxf(h2a[m][r] + bias2, 0.0f);
      }
  }
  __syncthreads();

  // score = h2 @ W3^T + b3 ; apply validity mask
  if (t < 64) {
    float s = b3[0];
#pragma unroll 8
    for (int k = 0; k < 64; ++k) s += (float)h2Lds[t * 64 + k] * W3[k];
    bool ok = (svm[(size_t)b * NS + s0 + t] != 0) && (cntLds[t] > 0);
    out[(size_t)b * NS + s0 + t] = ok ? s : -1000000000.0f;
  }
}

// ---------------------------------------------------------------------------
extern "C" void kernel_launch(void* const* d_in, const int* in_sizes, int n_in,
                              void* d_out, int out_size, void* d_ws, size_t ws_size,
                              hipStream_t stream) {
  (void)in_sizes; (void)n_in; (void)out_size; (void)ws_size;
  const float* nodef = (const float*)d_in[0];
  const float* gf    = (const float*)d_in[1];
  const unsigned char* scm = (const unsigned char*)d_in[2];
  const unsigned char* svm = (const unsigned char*)d_in[3];
  // d_in[4] cube_mask (all true) and d_in[5] batch (block-sorted) are implied
  const float* W1 = (const float*)d_in[6];
  const float* b1 = (const float*)d_in[7];
  const float* W2 = (const float*)d_in[8];
  const float* b2 = (const float*)d_in[9];
  const float* W3 = (const float*)d_in[10];
  const float* b3 = (const float*)d_in[11];
  float* out = (float*)d_out;

  // workspace layout
  _Float16* featsT = (_Float16*)d_ws;                        // 64*128*4096 f16
  _Float16* W1h    = featsT + (size_t)NB * ND * NC;          // 128*256 f16
  _Float16* W2h    = W1h + 128 * 256;                        // 64*128 f16
  float*    gW1b   = (float*)(W2h + 64 * 128);               // 64*128 f32

  prep_transpose<<<dim3(NB * (NC / 64)), dim3(256), 0, stream>>>(nodef, featsT);
  prep_weights<<<dim3((128 * 256 + 64 * 128) / 256), dim3(256), 0, stream>>>(W1, W2, W1h, W2h);
  prep_gw1<<<dim3(NB), dim3(128), 0, stream>>>(gf, W1, b1, gW1b);
  sep_head_main<<<dim3(NB * (NS / 64)), dim3(256), 0, stream>>>(
      scm, svm, featsT, W1h, W2h, gW1b, b2, W3, b3, out);
}
